// Block_74474732913330
// MI455X (gfx1250) — matrix operand order repaired and checked by reading, not compiled, not run on hardware
//
#include <hip/hip_runtime.h>
#include <hip/hip_bf16.h>

// ---------------- constants for this problem ----------------
#define B_ 4
#define T_ 1024
#define C_ 1024
#define H_ 16
#define N_ 64
#define FFN_ 4096
#define BT_ (B_ * T_)
#define BTC_ ((size_t)B_ * T_ * C_)
#define BTF_ ((size_t)B_ * T_ * FFN_)
#define EPS_ 1e-5f

// ---------------- vector types ----------------
typedef __attribute__((ext_vector_type(16))) __bf16 v16bf;
typedef __attribute__((ext_vector_type(8)))  __bf16 v8bf;
typedef __attribute__((ext_vector_type(8)))  float  v8f;
typedef unsigned int u32x4 __attribute__((ext_vector_type(4)));
typedef int          i32x8 __attribute__((ext_vector_type(8)));
typedef int          i32x4 __attribute__((ext_vector_type(4)));

// ---------------- helpers ----------------
__device__ __forceinline__ unsigned short f32_to_bf16_rne(float f) {
    union { float f; unsigned u; } c; c.f = f;
    unsigned u = c.u;
    unsigned rnd = 0x7FFFu + ((u >> 16) & 1u);
    u += rnd;
    return (unsigned short)(u >> 16);
}

__device__ __forceinline__ float block_reduce_sum(float v, float* sbuf, int n) {
    int tid = threadIdx.x;
    sbuf[tid] = v;
    __syncthreads();
    for (int s = n >> 1; s > 0; s >>= 1) {
        if (tid < s) sbuf[tid] += sbuf[tid + s];
        __syncthreads();
    }
    float r = sbuf[0];
    __syncthreads();
    return r;
}

// ---------------- Tensor Data Mover: 2D tile -> LDS ----------------
// D# per CDNA5 ISA ch.8: group0 = {count, lds_addr, global_addr, type=2},
// group1 = {data_size=2B, tensor_dim0=K, tensor_dim1=rows, tile_dim0=32,
//           tile_dim1=rows, tensor_dim0_stride=K}.  2D => groups 2/3 zero.
__device__ __forceinline__ void tdm_load_tile_2d(
    const unsigned short* gptr, unsigned lds_off, int rows, int kstride) {
    unsigned long long ga = (unsigned long long)gptr;
    u32x4 g0;
    g0[0] = 1u;                                            // count=1 (valid)
    g0[1] = lds_off;                                       // lds_addr (bytes)
    g0[2] = (unsigned)(ga & 0xFFFFFFFFu);                  // global_addr lo
    g0[3] = (unsigned)((ga >> 32) & 0x1FFFFFFu) | (2u << 30); // addr hi | type=2
    i32x8 g1;
    g1[0] = 0x00010000;                                    // data_size=2 bytes
    g1[1] = (int)((kstride & 0xFFFF) << 16);               // tensor_dim0[15:0]
    g1[2] = (int)(((unsigned)kstride >> 16) | (((unsigned)rows & 0xFFFFu) << 16));
    g1[3] = (int)(32u << 16);                              // tile_dim0 = 32
    g1[4] = rows;                                          // tile_dim1
    g1[5] = kstride;                                       // dim0 stride lo32
    g1[6] = 0;
    g1[7] = 0;
    i32x4 z4 = {};
#if defined(__clang_major__) && __clang_major__ >= 23
    i32x8 z8 = {};
    __builtin_amdgcn_tensor_load_to_lds(g0, g1, z4, z4, z8, 0);
#else
    __builtin_amdgcn_tensor_load_to_lds(g0, g1, z4, z4, 0);
#endif
}

// flat (generic) shared pointer -> 32-bit LDS byte offset (addr[31:0] in the
// LDS aperture is the workgroup-relative offset)
__device__ __forceinline__ unsigned lds_offset(const void* p) {
    return (unsigned)(unsigned long long)p;
}

// ---------------- f32 -> bf16 conversion ----------------
__global__ __launch_bounds__(256) void f32_to_bf16_kernel(
    const float* __restrict__ in, unsigned short* __restrict__ out, size_t n) {
    size_t i = (size_t)blockIdx.x * blockDim.x + threadIdx.x;
    if (i < n) out[i] = f32_to_bf16_rne(in[i]);
}

// ---------------- layernorm over last dim (C) ----------------
__global__ __launch_bounds__(256) void layernorm_kernel(
    const float* __restrict__ x, const float* __restrict__ w,
    const float* __restrict__ b, float* __restrict__ out, int C) {
    __shared__ float sbuf[256];
    int row = blockIdx.x;
    const float* xr = x + (size_t)row * C;
    float* orow = out + (size_t)row * C;
    float s1 = 0.f, s2 = 0.f;
    for (int c = threadIdx.x; c < C; c += 256) {
        float v = xr[c];
        s1 += v;
        s2 += v * v;
    }
    s1 = block_reduce_sum(s1, sbuf, 256);
    s2 = block_reduce_sum(s2, sbuf, 256);
    float mean = s1 / (float)C;
    float var  = s2 / (float)C - mean * mean;
    float inv  = rsqrtf(var + EPS_);
    for (int c = threadIdx.x; c < C; c += 256)
        orow[c] = (xr[c] - mean) * inv * w[c] + b[c];
}

// ---------------- time-shift token mixing -> bf16 outputs ----------------
template <int NM>
__global__ __launch_bounds__(256) void mix_kernel(
    const float* __restrict__ xln,
    const float* __restrict__ m0, const float* __restrict__ m1,
    const float* __restrict__ m2, const float* __restrict__ m3,
    unsigned short* __restrict__ o0, unsigned short* __restrict__ o1,
    unsigned short* __restrict__ o2, unsigned short* __restrict__ o3) {
    int row = blockIdx.x;           // b*T + t
    int t = row & (T_ - 1);
    const float* cur  = xln + (size_t)row * C_;
    const float* prev = xln + (size_t)(row - 1) * C_;
    for (int c = threadIdx.x; c < C_; c += 256) {
        float xc = cur[c];
        float xp = (t > 0) ? prev[c] : 0.f;
        size_t idx = (size_t)row * C_ + c;
        { float m = m0[c]; o0[idx] = f32_to_bf16_rne(xc * m + xp * (1.f - m)); }
        if (NM > 1) { float m = m1[c]; o1[idx] = f32_to_bf16_rne(xc * m + xp * (1.f - m)); }
        if (NM > 2) { float m = m2[c]; o2[idx] = f32_to_bf16_rne(xc * m + xp * (1.f - m)); }
        if (NM > 3) { float m = m3[c]; o3[idx] = f32_to_bf16_rne(xc * m + xp * (1.f - m)); }
    }
}

// ---------------- WMMA bf16 GEMM with TDM double-buffered LDS staging -----
// out = epilogue(A[M,K] @ W[N,K]^T), A/W bf16, acc f32.
// Block = 256 threads = 8 waves, tiled 4(M) x 2(N); block tile 128x128.
// Each wave owns a 32x64 register tile = 2 M-frags x 4 N-frags = 8 WMMAs
// per K-step (K-step 32), giving 8 WMMA per 12 ds_load_b128.
// Wave 0 drives the Tensor Data Mover: per K-step it DMAs the A tile
// (128x32) and W tile (128x32) into LDS (double buffered), hiding the next
// step's DMA under the current step's ds_load+WMMA, synchronized with
// s_wait_tensorcnt + workgroup barriers.
// Epilogues: 0=f32  1=silu->f32  2=acc+e0->f32  3=relu^2->bf16
//            4=sigmoid->f32  5=e0*acc+e1->f32
template <int EPI>
__global__ __launch_bounds__(256) void gemm_bf16_wmma(
    const unsigned short* __restrict__ A,
    const unsigned short* __restrict__ W,
    void* __restrict__ out,
    const float* __restrict__ e0,
    const float* __restrict__ e1,
    int M, int N, int K) {
    __shared__ __align__(16) unsigned short ldsA[2][128 * 32];
    __shared__ __align__(16) unsigned short ldsW[2][128 * 32];

    const int lane = threadIdx.x & 31;
    const int wave = threadIdx.x >> 5;
    const int wm = wave & 3;                    // 4 waves along M (32 rows each)
    const int wn = wave >> 2;                   // 2 waves along N (64 cols each)
    const int mBase = blockIdx.x * 128 + wm * 32;
    const int nBase = blockIdx.y * 128 + wn * 64;
    const int lm = lane & 15;
    const int hi = lane >> 4;

    const unsigned short* aTileG = A + (size_t)(blockIdx.x * 128) * K;
    const unsigned short* wTileG = W + (size_t)(blockIdx.y * 128) * K;

    v8f acc[2][4] = {};
    const int iters = K / 32;

    // prologue: DMA tiles for step 0 into buffer 0
    if (wave == 0) {
        tdm_load_tile_2d(aTileG, lds_offset(&ldsA[0][0]), 128, K);
        tdm_load_tile_2d(wTileG, lds_offset(&ldsW[0][0]), 128, K);
    }

    for (int i = 0; i < iters; ++i) {
        const int buf = i & 1;
        if (wave == 0) {
            if (i + 1 < iters) {
                // kick off next step's tiles into the other buffer, then wait
                // until only those two DMAs remain in flight
                tdm_load_tile_2d(aTileG + (i + 1) * 32,
                                 lds_offset(&ldsA[buf ^ 1][0]), 128, K);
                tdm_load_tile_2d(wTileG + (i + 1) * 32,
                                 lds_offset(&ldsW[buf ^ 1][0]), 128, K);
                __builtin_amdgcn_s_wait_tensorcnt(2);
            } else {
                __builtin_amdgcn_s_wait_tensorcnt(0);
            }
        }
        __syncthreads();

        // --- A fragments (16x32): lanes 0-15 hold K[0..7],[16..23];
        //     lanes 16-31 hold K[8..15],[24..31]
        v16bf afrag[2];
#pragma unroll
        for (int mi = 0; mi < 2; ++mi) {
            const unsigned short* aRow = &ldsA[buf][(wm * 32 + mi * 16 + lm) * 32];
            v8bf a_lo = *(const v8bf*)(aRow + hi * 8);
            v8bf a_hi = *(const v8bf*)(aRow + 16 + hi * 8);
#pragma unroll
            for (int e = 0; e < 8; ++e) { afrag[mi][e] = a_lo[e]; afrag[mi][8 + e] = a_hi[e]; }
        }

        // --- B fragments (32x16): lane n holds K[hi*8 .. hi*8+7],[16+hi*8 .. 16+hi*8+7] of col n
        v16bf bfrag[4];
#pragma unroll
        for (int j = 0; j < 4; ++j) {
            const unsigned short* wrow = &ldsW[buf][(wn * 64 + j * 16 + lm) * 32];
            v8bf b_lo = *(const v8bf*)(wrow + hi * 8);
            v8bf b_hi = *(const v8bf*)(wrow + 16 + hi * 8);
#pragma unroll
            for (int e = 0; e < 8; ++e) { bfrag[j][e] = b_lo[e]; bfrag[j][8 + e] = b_hi[e]; }
        }

#pragma unroll
        for (int mi = 0; mi < 2; ++mi)
#pragma unroll
            for (int j = 0; j < 4; ++j)
                acc[mi][j] = __builtin_amdgcn_wmma_f32_16x16x32_bf16(
                    false, afrag[mi], false, bfrag[j], (short)0, acc[mi][j], false, false);

        __syncthreads();
    }

    // --- epilogue + store. C/D layout: lanes 0-15: (m=g, n=lane);
    //     lanes 16-31: (m=8+g, n=lane-16).
#pragma unroll
    for (int mi = 0; mi < 2; ++mi) {
#pragma unroll
        for (int j = 0; j < 4; ++j) {
#pragma unroll
            for (int g = 0; g < 8; ++g) {
                int m = mBase + mi * 16 + hi * 8 + g;
                int n = nBase + j * 16 + lm;
                size_t idx = (size_t)m * N + n;
                float v = acc[mi][j][g];
                if (EPI == 0) {
                    ((float*)out)[idx] = v;
                } else if (EPI == 1) {                       // silu
                    ((float*)out)[idx] = v / (1.f + __expf(-v));
                } else if (EPI == 2) {                       // residual add
                    ((float*)out)[idx] = v + e0[idx];
                } else if (EPI == 3) {                       // relu^2 -> bf16
                    float t = fmaxf(v, 0.f);
                    ((unsigned short*)out)[idx] = f32_to_bf16_rne(t * t);
                } else if (EPI == 4) {                       // sigmoid
                    ((float*)out)[idx] = 1.f / (1.f + __expf(-v));
                } else {                                     // e0*acc + e1
                    ((float*)out)[idx] = fmaf(e0[idx], v, e1[idx]);
                }
            }
        }
    }
}

// ---------------- WKV5 recurrent scan ----------------
// One 64-thread block per (b,h). Thread j holds state column S[:,j] in VGPRs.
__global__ __launch_bounds__(64) void wkv_scan_kernel(
    const float* __restrict__ r, const float* __restrict__ k,
    const float* __restrict__ v, const float* __restrict__ decay,
    const float* __restrict__ u, float* __restrict__ y) {
    int b = blockIdx.x / H_;
    int h = blockIdx.x % H_;
    int j = threadIdx.x;

    __shared__ float ew_s[N_], u_s[N_], r_s[N_], k_s[N_];
    ew_s[j] = __expf(-__expf(decay[h * N_ + j]));
    u_s[j]  = u[h * N_ + j];

    float S[N_];
#pragma unroll
    for (int i = 0; i < N_; ++i) S[i] = 0.f;

    const size_t base = (size_t)b * T_ * C_ + h * N_;
    for (int t = 0; t < T_; ++t) {
        size_t idx = base + (size_t)t * C_;
        __syncthreads();
        r_s[j] = r[idx + j];
        k_s[j] = k[idx + j];
        __syncthreads();
        float vj = v[idx + j];
        float acc = 0.f;
#pragma unroll
        for (int i = 0; i < N_; ++i) {
            float kv = k_s[i] * vj;
            acc += r_s[i] * fmaf(u_s[i], kv, S[i]);
            S[i] = fmaf(ew_s[i], S[i], kv);
        }
        y[idx + j] = acc;
    }
}

// ---------------- groupnorm(y/8) * g -> bf16 ----------------
__global__ __launch_bounds__(64) void gn_gate_kernel(
    const float* __restrict__ y, const float* __restrict__ g,
    const float* __restrict__ lnx_w, const float* __restrict__ lnx_b,
    unsigned short* __restrict__ out) {
    __shared__ float sbuf[64];
    int blk = blockIdx.x;            // 0 .. B*T*H-1
    int h  = blk & (H_ - 1);
    int bt = blk >> 4;
    int tid = threadIdx.x;           // 0..63
    size_t idx = (size_t)bt * C_ + h * N_ + tid;

    float v = y[idx] * 0.125f;       // / HEAD_DIV
    float s1 = block_reduce_sum(v, sbuf, 64);
    float mean = s1 * (1.f / 64.f);
    float d = v - mean;
    float s2 = block_reduce_sum(d * d, sbuf, 64);
    float var = s2 * (1.f / 64.f);
    float xn = d * rsqrtf(var + EPS_);
    float o = xn * lnx_w[h * N_ + tid] + lnx_b[h * N_ + tid];
    out[idx] = f32_to_bf16_rne(o * g[idx]);
}

// ---------------- launch ----------------
extern "C" void kernel_launch(void* const* d_in, const int* in_sizes, int n_in,
                              void* d_out, int out_size, void* d_ws, size_t ws_size,
                              hipStream_t stream) {
    const float* x        = (const float*)d_in[0];
    const float* ln1_w    = (const float*)d_in[1];
    const float* ln1_b    = (const float*)d_in[2];
    const float* ln2_w    = (const float*)d_in[3];
    const float* ln2_b    = (const float*)d_in[4];
    const float* tm_mix_k = (const float*)d_in[5];
    const float* tm_mix_v = (const float*)d_in[6];
    const float* tm_mix_r = (const float*)d_in[7];
    const float* tm_mix_g = (const float*)d_in[8];
    const float* tm_decay = (const float*)d_in[9];
    const float* tm_u     = (const float*)d_in[10];
    const float* Wr       = (const float*)d_in[11];
    const float* Wk       = (const float*)d_in[12];
    const float* Wv       = (const float*)d_in[13];
    const float* Wg       = (const float*)d_in[14];
    const float* Wo       = (const float*)d_in[15];
    const float* lnx_w    = (const float*)d_in[16];
    const float* lnx_b    = (const float*)d_in[17];
    const float* cm_mix_k = (const float*)d_in[18];
    const float* cm_mix_r = (const float*)d_in[19];
    const float* cm_Wk    = (const float*)d_in[20];
    const float* cm_Wr    = (const float*)d_in[21];
    const float* cm_Wv    = (const float*)d_in[22];

    // workspace carve-up
    char* ws = (char*)d_ws;
    size_t off = 0;
    auto alloc = [&](size_t bytes) -> void* {
        off = (off + 255) & ~(size_t)255;
        void* p = ws + off;
        off += bytes;
        return p;
    };
    const size_t CC = (size_t)C_ * C_;
    const size_t CF = (size_t)C_ * FFN_;

    unsigned short* wr_bf   = (unsigned short*)alloc(CC * 2);
    unsigned short* wk_bf   = (unsigned short*)alloc(CC * 2);
    unsigned short* wv_bf   = (unsigned short*)alloc(CC * 2);
    unsigned short* wg_bf   = (unsigned short*)alloc(CC * 2);
    unsigned short* wo_bf   = (unsigned short*)alloc(CC * 2);
    unsigned short* cwr_bf  = (unsigned short*)alloc(CC * 2);
    unsigned short* cwk_bf  = (unsigned short*)alloc(CF * 2);
    unsigned short* cwv_bf  = (unsigned short*)alloc(CF * 2);
    float*          xln     = (float*)alloc(BTC_ * 4);
    unsigned short* xk_bf   = (unsigned short*)alloc(BTC_ * 2);
    unsigned short* xv_bf   = (unsigned short*)alloc(BTC_ * 2);
    unsigned short* xr_bf   = (unsigned short*)alloc(BTC_ * 2);
    unsigned short* xg_bf   = (unsigned short*)alloc(BTC_ * 2);
    float*          r_f     = (float*)alloc(BTC_ * 4);
    float*          k_f     = (float*)alloc(BTC_ * 4);
    float*          v_f     = (float*)alloc(BTC_ * 4);
    float*          g_f     = (float*)alloc(BTC_ * 4);
    float*          y_f     = (float*)alloc(BTC_ * 4);
    unsigned short* yg_bf   = (unsigned short*)alloc(BTC_ * 2);
    float*          x2_f    = (float*)alloc(BTC_ * 4);
    float*          rr_f    = (float*)alloc(BTC_ * 4);
    unsigned short* kk_bf   = (unsigned short*)alloc(BTF_ * 2);
    (void)ws_size; (void)in_sizes; (void)n_in; (void)out_size;

    // 1) weights -> bf16 (L2-resident for the GEMMs)
    {
        int thr = 256;
        f32_to_bf16_kernel<<<(CC + thr - 1) / thr, thr, 0, stream>>>(Wr, wr_bf, CC);
        f32_to_bf16_kernel<<<(CC + thr - 1) / thr, thr, 0, stream>>>(Wk, wk_bf, CC);
        f32_to_bf16_kernel<<<(CC + thr - 1) / thr, thr, 0, stream>>>(Wv, wv_bf, CC);
        f32_to_bf16_kernel<<<(CC + thr - 1) / thr, thr, 0, stream>>>(Wg, wg_bf, CC);
        f32_to_bf16_kernel<<<(CC + thr - 1) / thr, thr, 0, stream>>>(Wo, wo_bf, CC);
        f32_to_bf16_kernel<<<(CC + thr - 1) / thr, thr, 0, stream>>>(cm_Wr, cwr_bf, CC);
        f32_to_bf16_kernel<<<(CF + thr - 1) / thr, thr, 0, stream>>>(cm_Wk, cwk_bf, CF);
        f32_to_bf16_kernel<<<(CF + thr - 1) / thr, thr, 0, stream>>>(cm_Wv, cwv_bf, CF);
    }

    // 2) ln1(x) and time-shift mixes (bf16 GEMM inputs)
    layernorm_kernel<<<BT_, 256, 0, stream>>>(x, ln1_w, ln1_b, xln, C_);
    mix_kernel<4><<<BT_, 256, 0, stream>>>(xln, tm_mix_k, tm_mix_v, tm_mix_r, tm_mix_g,
                                           xk_bf, xv_bf, xr_bf, xg_bf);

    // 3) projection GEMMs (M=4096, N=1024, K=1024)
    dim3 gemm_grid_c(BT_ / 128, C_ / 128);
    gemm_bf16_wmma<0><<<gemm_grid_c, 256, 0, stream>>>(xr_bf, wr_bf, r_f, nullptr, nullptr, BT_, C_, C_);
    gemm_bf16_wmma<0><<<gemm_grid_c, 256, 0, stream>>>(xk_bf, wk_bf, k_f, nullptr, nullptr, BT_, C_, C_);
    gemm_bf16_wmma<0><<<gemm_grid_c, 256, 0, stream>>>(xv_bf, wv_bf, v_f, nullptr, nullptr, BT_, C_, C_);
    gemm_bf16_wmma<1><<<gemm_grid_c, 256, 0, stream>>>(xg_bf, wg_bf, g_f, nullptr, nullptr, BT_, C_, C_);

    // 4) sequential WKV scan: one block per (b,h)
    wkv_scan_kernel<<<B_ * H_, 64, 0, stream>>>(r_f, k_f, v_f, tm_decay, tm_u, y_f);

    // 5) groupnorm(y/8)*silu(g) -> bf16
    gn_gate_kernel<<<BT_ * H_, 64, 0, stream>>>(y_f, g_f, lnx_w, lnx_b, yg_bf);

    // 6) output projection + residual: x2 = x + yg @ Wo^T
    gemm_bf16_wmma<2><<<gemm_grid_c, 256, 0, stream>>>(yg_bf, wo_bf, x2_f, x, nullptr, BT_, C_, C_);

    // 7) channel mix: ln2, mixes
    layernorm_kernel<<<BT_, 256, 0, stream>>>(x2_f, ln2_w, ln2_b, xln, C_);
    mix_kernel<2><<<BT_, 256, 0, stream>>>(xln, cm_mix_k, cm_mix_r, nullptr, nullptr,
                                           xk_bf, xr_bf, nullptr, nullptr);

    // 8) kk = relu(xk @ cm_Wk^T)^2 -> bf16   (M=4096, N=4096, K=1024)
    dim3 gemm_grid_f(BT_ / 128, FFN_ / 128);
    gemm_bf16_wmma<3><<<gemm_grid_f, 256, 0, stream>>>(xk_bf, cwk_bf, kk_bf, nullptr, nullptr, BT_, FFN_, C_);

    // 9) rr = sigmoid(xr @ cm_Wr^T)
    gemm_bf16_wmma<4><<<gemm_grid_c, 256, 0, stream>>>(xr_bf, cwr_bf, rr_f, nullptr, nullptr, BT_, C_, C_);

    // 10) out = x2 + rr * (kk @ cm_Wv^T)     (M=4096, N=1024, K=4096)
    gemm_bf16_wmma<5><<<gemm_grid_c, 256, 0, stream>>>(kk_bf, cwv_bf, (float*)d_out, rr_f, x2_f, BT_, C_, FFN_);
}
